// RelGATLayer_72035191489140
// MI455X (gfx1250) — matrix operand order repaired and checked by reading, not compiled, not run on hardware
//
#include <hip/hip_runtime.h>
#include <stdint.h>

#define N_NODES 50000
#define N_EDGES 800000
#define DIM 128
#define REL_SWAP 17
#define SLOPE 0.01f
#define WSTRIDE 136   // bf16 weight rows: 272B stride (16B aligned, bank-spread)
#define RSTRIDE 132   // f32 rel-tile rows: 528B stride via TDM pad (16B aligned, bank-spread)

typedef __attribute__((ext_vector_type(16))) __bf16 v16bf;
typedef __attribute__((ext_vector_type(8)))  float  v8f;
typedef __attribute__((ext_vector_type(4)))  unsigned int v4u;
typedef __attribute__((ext_vector_type(8)))  unsigned int v8u;

union Frag16 { v16bf v; __bf16 b[16]; unsigned short u[16]; uint4 q[2]; };
union BfBits { __bf16 h; unsigned short u; };

__device__ __forceinline__ unsigned short f2bfu(float f) {
  BfBits c; c.h = (__bf16)f; return c.u;       // native RNE convert (v_cvt_pk_bf16_f32)
}

// order-preserving float->uint encoding for atomic segment-max
__device__ __forceinline__ unsigned fenc(float f) {
  unsigned u = __float_as_uint(f);
  return (u & 0x80000000u) ? ~u : (u | 0x80000000u);
}
__device__ __forceinline__ float fdec(unsigned e) {
  unsigned u = (e & 0x80000000u) ? (e & 0x7FFFFFFFu) : ~e;
  return __uint_as_float(u);
}

// Stage a 128x128 row-major f32 weight into LDS as bf16, transposed to [n][k]
__device__ __forceinline__ void stage_w(const float* __restrict__ W, unsigned short* Wlds) {
  for (int i = threadIdx.x; i < DIM * DIM; i += blockDim.x) {
    int k = i >> 7, n = i & 127;
    Wlds[n * WSTRIDE + k] = f2bfu(W[i]);
  }
}

// A fragment (16x32 bf16): lane half h, row per lane; K = kk*32 + h*8 + {0..7} and +16
__device__ __forceinline__ v16bf load_a_scaled(const float* __restrict__ rowp, int kk, int half, float s) {
  const int c0 = kk * 32 + half * 8;
  const float4* pa = (const float4*)(rowp + c0);
  const float4* pb = (const float4*)(rowp + c0 + 16);
  float4 x0 = pa[0], x1 = pa[1], x2 = pb[0], x3 = pb[1];
  Frag16 a;
  a.b[0]  = (__bf16)(x0.x * s); a.b[1]  = (__bf16)(x0.y * s); a.b[2]  = (__bf16)(x0.z * s); a.b[3]  = (__bf16)(x0.w * s);
  a.b[4]  = (__bf16)(x1.x * s); a.b[5]  = (__bf16)(x1.y * s); a.b[6]  = (__bf16)(x1.z * s); a.b[7]  = (__bf16)(x1.w * s);
  a.b[8]  = (__bf16)(x2.x * s); a.b[9]  = (__bf16)(x2.y * s); a.b[10] = (__bf16)(x2.z * s); a.b[11] = (__bf16)(x2.w * s);
  a.b[12] = (__bf16)(x3.x * s); a.b[13] = (__bf16)(x3.y * s); a.b[14] = (__bf16)(x3.z * s); a.b[15] = (__bf16)(x3.w * s);
  return a.v;
}

// B fragment (32x16 bf16): lane = column, K = kk*32 + h*8 + {0..7} and +16 from transposed LDS weight
__device__ __forceinline__ v16bf load_b(const unsigned short* Wlds, int col, int kk, int half) {
  Frag16 b;
  const uint4* p = (const uint4*)(Wlds + col * WSTRIDE + kk * 32 + half * 8);
  b.q[0] = p[0]; b.q[1] = p[2];
  return b.v;
}

#define WMMA_BF16(c, a, b) \
  __builtin_amdgcn_wmma_f32_16x16x32_bf16(false, (a), false, (b), (short)0, (c), false, false)

// ---------------- kernel 0: init segment state ----------------
__global__ void k_zero(unsigned* menc, float* den, float* num) {
  int i = blockIdx.x * blockDim.x + threadIdx.x;
  if (i < N_NODES) { menc[i] = 0u; den[i] = 0.f; }
  size_t stride = (size_t)gridDim.x * blockDim.x;
  for (size_t j = i; j < (size_t)N_NODES * DIM; j += stride) num[j] = 0.f;
}

// ---------------- kernel 1: p1 = ent@W1, p2 = ent@W2 ----------------
__global__ __launch_bounds__(256) void k_proj(const float* __restrict__ ent,
                                              const float* __restrict__ Wn,
                                              float* __restrict__ p1, float* __restrict__ p2) {
  __shared__ __align__(16) unsigned short Wlds[DIM * WSTRIDE];
  const int lane = threadIdx.x & 31, wave = threadIdx.x >> 5;
  const int half = lane >> 4, l16 = lane & 15;
  const int rowbase = blockIdx.x * 128 + wave * 16;
  int row = rowbase + l16; if (row > N_NODES - 1) row = N_NODES - 1;
  const float* arow = ent + (size_t)row * DIM;

  for (int m = 0; m < 2; ++m) {
    __syncthreads();
    stage_w(Wn + (size_t)m * DIM * DIM, Wlds);
    __syncthreads();
    v8f c[8] = {};
#pragma unroll
    for (int kk = 0; kk < 4; ++kk) {
      v16bf a = load_a_scaled(arow, kk, half, 1.0f);
#pragma unroll
      for (int nt = 0; nt < 8; ++nt) {
        v16bf b = load_b(Wlds, nt * 16 + l16, kk, half);
        c[nt] = WMMA_BF16(c[nt], a, b);
      }
    }
    float* out = m ? p2 : p1;
#pragma unroll
    for (int g = 0; g < 8; ++g) {
      int r = rowbase + g + 8 * half;
      if (r < N_NODES) {
#pragma unroll
        for (int nt = 0; nt < 8; ++nt) out[(size_t)r * DIM + nt * 16 + l16] = c[nt][g];
      }
    }
  }
}

// ---------------- kernel 2: att = rel@W3 + gathered p rows; scores; segment max ----------------
// A-matrix (128 consecutive edge rows, 64KB contiguous) staged by the Tensor Data Mover,
// with TDM padding (4 DWORDs per 128 DWORDs) to de-conflict LDS banks.
__global__ __launch_bounds__(256) void k_edge(const float* __restrict__ rel,
                                              const float* __restrict__ Wn,
                                              const float* __restrict__ avec,
                                              const float* __restrict__ p1,
                                              const float* __restrict__ p2,
                                              const int* __restrict__ src,
                                              const int* __restrict__ dst,
                                              const int* __restrict__ rid,
                                              float* __restrict__ att,
                                              float* __restrict__ score,
                                              unsigned* __restrict__ menc) {
  __shared__ __align__(16) unsigned short Wlds[DIM * WSTRIDE];
  __shared__ float a_sh[DIM];
  __shared__ __align__(16) float relT[128 * RSTRIDE];

  const int ebase = blockIdx.x * 128;

  if (threadIdx.x < 32) {
    // Tensor DMA descriptor (D#): 2D tensor, 128x128 f32 tile, row padding 4 DWORDs/row
    unsigned long long ga = (unsigned long long)(const void*)(rel + (size_t)ebase * DIM);
    unsigned ldsoff = (unsigned)(size_t)(&relT[0]);   // low 32 bits of LDS aperture addr = LDS offset
    v4u g0;
    g0[0] = 1u;                                            // count=1 valid descriptor
    g0[1] = ldsoff;                                        // lds_addr
    g0[2] = (unsigned)ga;                                  // global_addr[31:0]
    g0[3] = ((unsigned)(ga >> 32) & 0x1FFFFFFu) | (2u << 30); // global_addr[56:32] | type=2
    v8u g1;
    g1[0] = (2u << 16)          // data_size = 4B
          | (1u << 20)          // pad_enable
          | (6u << 22)          // pad_interval: every 128 DWORDs
          | (3u << 25);         // pad_amount: 4 DWORDs
    g1[1] = 128u << 16;         // tensor_dim0[15:0]
    g1[2] = 128u << 16;         // tensor_dim1[15:0]
    g1[3] = 128u << 16;         // tile_dim0
    g1[4] = 128u;               // tile_dim1
    g1[5] = 128u;               // tensor_dim0_stride[31:0]
    g1[6] = 0u;
    g1[7] = 0u;
    asm volatile("tensor_load_to_lds %0, %1" :: "s"(g0), "s"(g1) : "memory");
    __builtin_amdgcn_s_wait_tensorcnt(0);
  }
  stage_w(Wn + 2 * DIM * DIM, Wlds);
  for (int i = threadIdx.x; i < DIM; i += blockDim.x) a_sh[i] = avec[i];
  __syncthreads();

  const int lane = threadIdx.x & 31, wave = threadIdx.x >> 5;
  const int half = lane >> 4, l16 = lane & 15;
  const float* arow = relT + (wave * 16 + l16) * RSTRIDE;   // LDS-resident A row

  v8f c[8] = {};
#pragma unroll
  for (int kk = 0; kk < 4; ++kk) {
    v16bf a = load_a_scaled(arow, kk, half, 1.0f);
#pragma unroll
    for (int nt = 0; nt < 8; ++nt) {
      v16bf b = load_b(Wlds, nt * 16 + l16, kk, half);
      c[nt] = WMMA_BF16(c[nt], a, b);
    }
  }

#pragma unroll
  for (int g = 0; g < 8; ++g) {
    int er = ebase + wave * 16 + g + 8 * half;   // E % 128 == 0: always in range
    int s = src[er], d = dst[er];
    bool swap = rid[er] >= REL_SWAP;
    const float* q1 = p1 + (size_t)(swap ? d : s) * DIM;
    const float* q2 = p2 + (size_t)(swap ? s : d) * DIM;
    float vals[8];
    float part = 0.f;
#pragma unroll
    for (int nt = 0; nt < 8; ++nt) {
      int col = nt * 16 + l16;
      float v = c[nt][g] + q1[col] + q2[col];
      vals[nt] = v;
      part += v * a_sh[col];
    }
    // reduce over the 16 lanes of this half (halves hold different rows)
    part += __shfl_xor(part, 1, 32);
    part += __shfl_xor(part, 2, 32);
    part += __shfl_xor(part, 4, 32);
    part += __shfl_xor(part, 8, 32);
    float* orow = att + (size_t)er * DIM;
#pragma unroll
    for (int nt = 0; nt < 8; ++nt) orow[nt * 16 + l16] = vals[nt];
    if (l16 == 0) {
      float sc = part > 0.f ? part : part * SLOPE;
      score[er] = sc;
      atomicMax(menc + d, fenc(sc));
    }
  }
}

// ---------------- kernel 3: softmax numerator/denominator (L2-resident atomics) ----------------
__global__ __launch_bounds__(256) void k_soft(const float* __restrict__ att,
                                              const float* __restrict__ score,
                                              const int* __restrict__ dst,
                                              const unsigned* __restrict__ menc,
                                              float* __restrict__ den,
                                              float* __restrict__ num) {
  const int lane = threadIdx.x & 31;
  const int e = blockIdx.x * (blockDim.x >> 5) + (threadIdx.x >> 5);
  if (e >= N_EDGES) return;
  int d = dst[e];
  float ex = __expf(score[e] - fdec(menc[d]));
  if (lane == 0) atomicAdd(den + d, ex);
  const float* arow = att + (size_t)e * DIM;
  float* nrow = num + (size_t)d * DIM;
#pragma unroll
  for (int j = 0; j < 4; ++j) {
    int col = lane + j * 32;
    atomicAdd(nrow + col, ex * arow[col]);
  }
}

// ---------------- kernel 4: node_h = leaky_relu([ent | num/den] @ Wt + bt) ----------------
__global__ __launch_bounds__(256) void k_out(const float* __restrict__ ent,
                                             const float* __restrict__ num,
                                             const float* __restrict__ den,
                                             const float* __restrict__ Wt,
                                             const float* __restrict__ bt,
                                             float* __restrict__ out) {
  __shared__ __align__(16) unsigned short Wlds[DIM * WSTRIDE];
  __shared__ float bt_sh[DIM];
  const int lane = threadIdx.x & 31, wave = threadIdx.x >> 5;
  const int half = lane >> 4, l16 = lane & 15;
  const int rowbase = blockIdx.x * 128 + wave * 16;
  int row = rowbase + l16; if (row > N_NODES - 1) row = N_NODES - 1;

  for (int i = threadIdx.x; i < DIM; i += blockDim.x) bt_sh[i] = bt[i];

  v8f c[8] = {};

  // phase 0: ent @ Wt[0:128,:]
  stage_w(Wt, Wlds);
  __syncthreads();
  {
    const float* arow = ent + (size_t)row * DIM;
#pragma unroll
    for (int kk = 0; kk < 4; ++kk) {
      v16bf a = load_a_scaled(arow, kk, half, 1.0f);
#pragma unroll
      for (int nt = 0; nt < 8; ++nt) {
        v16bf b = load_b(Wlds, nt * 16 + l16, kk, half);
        c[nt] = WMMA_BF16(c[nt], a, b);
      }
    }
  }
  __syncthreads();

  // phase 1: h @ Wt[128:256,:], h = num/den (0 for isolated nodes), same accumulators
  stage_w(Wt + DIM * DIM, Wlds);
  __syncthreads();
  {
    float dv = den[row];
    float dinv = dv != 0.f ? 1.f / dv : 0.f;
    const float* arow = num + (size_t)row * DIM;
#pragma unroll
    for (int kk = 0; kk < 4; ++kk) {
      v16bf a = load_a_scaled(arow, kk, half, dinv);
#pragma unroll
      for (int nt = 0; nt < 8; ++nt) {
        v16bf b = load_b(Wlds, nt * 16 + l16, kk, half);
        c[nt] = WMMA_BF16(c[nt], a, b);
      }
    }
  }

#pragma unroll
  for (int g = 0; g < 8; ++g) {
    int r = rowbase + g + 8 * half;
    if (r < N_NODES) {
#pragma unroll
      for (int nt = 0; nt < 8; ++nt) {
        int col = nt * 16 + l16;
        float v = c[nt][g] + bt_sh[col];
        v = v > 0.f ? v : v * SLOPE;
        out[(size_t)r * DIM + col] = v;
      }
    }
  }
}

extern "C" void kernel_launch(void* const* d_in, const int* in_sizes, int n_in,
                              void* d_out, int out_size, void* d_ws, size_t ws_size,
                              hipStream_t stream) {
  (void)in_sizes; (void)n_in; (void)out_size; (void)ws_size;
  const float* ent  = (const float*)d_in[0];
  const float* rel  = (const float*)d_in[1];
  const int*   src  = (const int*)d_in[2];
  const int*   dst  = (const int*)d_in[3];
  const int*   rid  = (const int*)d_in[4];
  const float* Wn   = (const float*)d_in[5];
  const float* avec = (const float*)d_in[6];
  const float* Wt   = (const float*)d_in[7];
  const float* bt   = (const float*)d_in[8];
  float* out = (float*)d_out;

  // workspace layout (floats)
  float*    p1    = (float*)d_ws;                       // N*D
  float*    p2    = p1 + (size_t)N_NODES * DIM;         // N*D
  float*    num   = p2 + (size_t)N_NODES * DIM;         // N*D
  float*    den   = num + (size_t)N_NODES * DIM;        // N
  unsigned* menc  = (unsigned*)(den + N_NODES);         // N
  float*    score = (float*)(menc + N_NODES);           // E
  float*    att   = score + N_EDGES;                    // E*D (largest, last)

  // second output: verbatim rel_feats copy
  hipMemcpyAsync(out + (size_t)N_NODES * DIM, rel,
                 (size_t)N_EDGES * DIM * sizeof(float),
                 hipMemcpyDeviceToDevice, stream);

  k_zero<<<(N_NODES + 255) / 256, 256, 0, stream>>>(menc, den, num);
  k_proj<<<(N_NODES + 127) / 128, 256, 0, stream>>>(ent, Wn, p1, p2);
  k_edge<<<N_EDGES / 128, 256, 0, stream>>>(rel, Wn, avec, p1, p2, src, dst, rid, att, score, menc);
  k_soft<<<N_EDGES / 8, 256, 0, stream>>>(att, score, dst, menc, den, num);
  k_out<<<(N_NODES + 127) / 128, 256, 0, stream>>>(ent, num, den, Wt, bt, out);
}
